// RWKV7Cell_66924180407318
// MI455X (gfx1250) — matrix operand order repaired and checked by reading, not compiled, not run on hardware
//
#include <hip/hip_runtime.h>
#include <hip/hip_bf16.h>
#include <cstdint>

#define TT 512
#define BB 8
#define CC 1024
#define HEADS 16
#define HEADDIM 64
#define MROWS (TT * BB) /* 4096 */

typedef __attribute__((ext_vector_type(16))) __bf16 bf16x16;
typedef __attribute__((ext_vector_type(8)))  float  f32x8;

union Frag16 { bf16x16 v; uint4 q[2]; };

__device__ __forceinline__ unsigned short f2bf(float f) {
  union { float f; uint32_t u; } x; x.f = f;
  uint32_t u = x.u;
  if ((u & 0x7F800000u) == 0x7F800000u && (u & 0x007FFFFFu)) {
    return (unsigned short)((u >> 16) | 0x0040u); // quiet the NaN
  }
  uint32_t r = u + 0x7FFFu + ((u >> 16) & 1u);    // round-to-nearest-even
  return (unsigned short)(r >> 16);
}

// -------------------------- fp32 weight (K x N) -> bf16 transposed (N x K)
__global__ __launch_bounds__(256) void w2bfT_kernel(
    const float* __restrict__ in, unsigned short* __restrict__ out) {
  __shared__ unsigned short tile[32][33];
  int bn = blockIdx.x * 32;   // n block
  int bk = blockIdx.y * 32;   // k block
  int tx = threadIdx.x, ty = threadIdx.y;   // (32, 8)
  #pragma unroll
  for (int i = 0; i < 32; i += 8)
    tile[ty + i][tx] = f2bf(in[(size_t)(bk + ty + i) * CC + bn + tx]); // [k][n]
  __syncthreads();
  #pragma unroll
  for (int i = 0; i < 32; i += 8)
    out[(size_t)(bn + ty + i) * CC + bk + tx] = tile[tx][ty + i];      // [n][k]
}

// ------------------------------------------------- token-shift mix (6 outputs)
__global__ void mix_kernel(const float* __restrict__ x,
                           const float* __restrict__ mr, const float* __restrict__ mw,
                           const float* __restrict__ mk, const float* __restrict__ mv,
                           const float* __restrict__ ma, const float* __restrict__ mg,
                           unsigned short* __restrict__ xr, unsigned short* __restrict__ xw,
                           unsigned short* __restrict__ xk, unsigned short* __restrict__ xv,
                           unsigned short* __restrict__ xa, unsigned short* __restrict__ xg) {
  int i = blockIdx.x * blockDim.x + threadIdx.x;
  if (i >= MROWS * CC) return;
  int c   = i & (CC - 1);
  int row = i >> 10;       // / CC
  int t   = row / BB;
  float xc = x[i];
  float xp = (t == 0) ? 0.f : x[i - BB * CC];
  float xx = xp - xc;
  xr[i] = f2bf(xc + xx * mr[c]);
  xw[i] = f2bf(xc + xx * mw[c]);
  xk[i] = f2bf(xc + xx * mk[c]);
  xv[i] = f2bf(xc + xx * mv[c]);
  xa[i] = f2bf(xc + xx * ma[c]);
  xg[i] = f2bf(xc + xx * mg[c]);
}

// --------------------------------------------------------- bf16 WMMA GEMM
// C[M=4096, N=1024] = A[M,K=1024](bf16, row-major) * B (given as BT[N,K] bf16).
// 256 threads / 8 waves, 128x128 tile, double-buffered LDS staged with
// global_load_async_to_lds_b128 (ASYNCcnt), s_wait_asynccnt + barrier sync.
__global__ __launch_bounds__(256) void gemm_bf16_kernel(
    const unsigned short* __restrict__ A,
    const unsigned short* __restrict__ BT,
    float* __restrict__ Co) {
  __shared__ unsigned short lA[2][128 * 32];   // [row][k]
  __shared__ unsigned short lB[2][128 * 32];   // [col][k]  (BT rows)
  const int tid  = threadIdx.x;
  const int lane = tid & 31, wv = tid >> 5;
  const int m0 = blockIdx.y * 128, n0 = blockIdx.x * 128;
  const int mbase = (wv & 3) * 32, nbase = (wv >> 2) * 64;
  const int lr = lane & 15, hf = lane >> 4;

  // stage one 128x32 K-tile of A and BT into LDS buffer `buf` (async DMA)
  auto stage = [&](int buf, int k0) {
    #pragma unroll
    for (int ch = tid; ch < 512; ch += 256) {
      int row = ch >> 2, k8 = (ch & 3) << 3;
      uint32_t la = (uint32_t)(uintptr_t)&lA[buf][row * 32 + k8];
      const unsigned short* ga = A + (size_t)(m0 + row) * CC + k0 + k8;
      asm volatile("global_load_async_to_lds_b128 %0, %1, off"
                   :: "v"(la), "v"(ga) : "memory");
      uint32_t lb = (uint32_t)(uintptr_t)&lB[buf][row * 32 + k8];
      const unsigned short* gb = BT + (size_t)(n0 + row) * CC + k0 + k8;
      asm volatile("global_load_async_to_lds_b128 %0, %1, off"
                   :: "v"(lb), "v"(gb) : "memory");
    }
  };

  f32x8 acc[2][4];
  #pragma unroll
  for (int i = 0; i < 2; i++)
    #pragma unroll
    for (int j = 0; j < 4; j++)
      #pragma unroll
      for (int v = 0; v < 8; v++) acc[i][j][v] = 0.f;

  stage(0, 0);
  asm volatile("s_wait_asynccnt 0" ::: "memory");
  __syncthreads();

  const int NT = CC / 32;
  for (int kt = 0; kt < NT; kt++) {
    const int cur = kt & 1;
    if (kt + 1 < NT) stage(cur ^ 1, (kt + 1) * 32);

    // A fragment (ISA 7.12.2, 16-bit A 16x32): lane half hf=0 -> elems 0-7 =
    // K0-7, elems 8-15 = K16-23 ; hf=1 -> K8-15 / K24-31.
    Frag16 af[2], bfr[4];
    #pragma unroll
    for (int i = 0; i < 2; i++) {
      const unsigned short* p = &lA[cur][(mbase + i * 16 + lr) * 32 + hf * 8];
      af[i].q[0] = *(const uint4*)p;
      af[i].q[1] = *(const uint4*)(p + 16);
    }
    // B fragment: same K order as A: hf=0 -> elems 0-7 = K0-7, elems 8-15 =
    // K16-23 ; hf=1 -> K8-15 / K24-31.
    #pragma unroll
    for (int j = 0; j < 4; j++) {
      const unsigned short* p = &lB[cur][(nbase + j * 16 + lr) * 32 + hf * 8];
      bfr[j].q[0] = *(const uint4*)p;
      bfr[j].q[1] = *(const uint4*)(p + 16);
    }
    #pragma unroll
    for (int i = 0; i < 2; i++)
      #pragma unroll
      for (int j = 0; j < 4; j++)
        acc[i][j] = __builtin_amdgcn_wmma_f32_16x16x32_bf16(
            false, af[i].v, false, bfr[j].v, (short)0, acc[i][j], false, false);

    asm volatile("s_wait_asynccnt 0" ::: "memory");
    __syncthreads();
  }

  // C/D layout: VGPR v holds (M = v + 8*hf, N = lr) of each 16x16 tile
  #pragma unroll
  for (int i = 0; i < 2; i++)
    #pragma unroll
    for (int j = 0; j < 4; j++) {
      int col = n0 + nbase + j * 16 + lr;
      #pragma unroll
      for (int v = 0; v < 8; v++) {
        int row = m0 + mbase + i * 16 + hf * 8 + v;
        Co[(size_t)row * CC + col] = acc[i][j][v];
      }
    }
}

// ----------------------------------------- decay/sigmoids/kk-normalize (per head)
__global__ void prep_kernel(float* __restrict__ wbuf,  // in raw -> out decay
                            float* __restrict__ kbuf,  // in raw -> out k'
                            float* __restrict__ abuf,  // in raw -> out sigmoid(a0+.)
                            float* __restrict__ gbuf,  // in raw -> out sigmoid
                            float* __restrict__ kkbuf, // out normalized kk
                            const float* __restrict__ w0, const float* __restrict__ a0,
                            const float* __restrict__ k_k, const float* __restrict__ k_a) {
  int id = blockIdx.x * blockDim.x + threadIdx.x;
  if (id >= MROWS * HEADS) return;
  int row = id / HEADS, h = id % HEADS;
  int base = row * CC + h * HEADDIM;
  int cb = h * HEADDIM;
  float ssq = 0.f;
  #pragma unroll 4
  for (int m = 0; m < HEADDIM; m++) {
    int c = cb + m, ix = base + m;
    float u = w0[c] + wbuf[ix];
    float wlog = -log1pf(expf(-u)) - 0.5f;      // -softplus(-u) - 0.5
    wbuf[ix] = expf(-expf(wlog));
    float a = 1.f / (1.f + expf(-(a0[c] + abuf[ix])));
    abuf[ix] = a;
    gbuf[ix] = 1.f / (1.f + expf(-gbuf[ix]));
    float kv  = kbuf[ix];
    float kkv = kv * k_k[c];
    kkbuf[ix] = kkv;
    ssq += kkv * kkv;
    kbuf[ix] = kv * (1.f + (a - 1.f) * k_a[c]);
  }
  float inv = 1.f / fmaxf(sqrtf(ssq), 1e-12f);
  #pragma unroll 4
  for (int m = 0; m < HEADDIM; m++) kkbuf[base + m] *= inv;
}

// ---------------------------------- sequential WKV7 scan, one block per (b,h)
__global__ __launch_bounds__(64) void scan_kernel(
    const float* __restrict__ rbuf, const float* __restrict__ wbuf,
    const float* __restrict__ kbuf, const float* __restrict__ vbuf,
    const float* __restrict__ abuf, const float* __restrict__ gbuf,
    const float* __restrict__ kkbuf,
    const float* __restrict__ r_k, const float* __restrict__ lnx_w,
    const float* __restrict__ lnx_b,
    unsigned short* __restrict__ yg) {
  int bh = blockIdx.x;
  int b = bh / HEADS, h = bh % HEADS;
  int m = threadIdx.x;
  int c = h * HEADDIM + m;
  float rk_m = r_k[c], lw = lnx_w[c], lb = lnx_b[c];
  __shared__ float sr[64], sw[64], sk[64], skk[64], skka[64], sv[64];
  __shared__ float rA[64], rB[64], rC[64];
  float S[64];
  #pragma unroll
  for (int n = 0; n < 64; n++) S[n] = 0.f;

  for (int t = 0; t < TT; t++) {
    int base = (t * BB + b) * CC + h * HEADDIM;
    float rm = rbuf[base + m], km = kbuf[base + m], kkm = kkbuf[base + m];
    float am = abuf[base + m], vm = vbuf[base + m], wm = wbuf[base + m];
    float gm = gbuf[base + m];
    sr[m] = rm; sw[m] = wm; sk[m] = km;
    skk[m] = kkm; skka[m] = kkm * am; sv[m] = vm;
    __syncthreads();

    float sa = 0.f;
    #pragma unroll
    for (int n = 0; n < 64; n++) sa += S[n] * skk[n];
    sa = -sa;

    float y = 0.f;
    #pragma unroll
    for (int n = 0; n < 64; n++) {
      float s = S[n] * sw[n] + sa * skka[n] + vm * sk[n];
      S[n] = s;
      y += s * sr[n];
    }
    rA[m] = y; rB[m] = y * y; rC[m] = rm * km * rk_m;
    __syncthreads();
    #pragma unroll
    for (int s = 32; s > 0; s >>= 1) {
      if (m < s) { rA[m] += rA[m + s]; rB[m] += rB[m + s]; rC[m] += rC[m + s]; }
      __syncthreads();
    }
    float mean = rA[0] * (1.f / 64.f);
    float var  = rB[0] * (1.f / 64.f) - mean * mean;
    float rks  = rC[0];
    float yn   = (y - mean) * rsqrtf(var + 64e-5f) * lw + lb;
    yg[base + m] = f2bf((yn + rks * vm) * gm);
    __syncthreads();
  }
}

// --------------------------------------- layernorm(inp + att) -> bf16 (for Wf)
__global__ __launch_bounds__(256) void ln_add_kernel(
    const float* __restrict__ inp, const float* __restrict__ att,
    const float* __restrict__ nw, const float* __restrict__ nb,
    unsigned short* __restrict__ hout) {
  int row = blockIdx.x, tid = threadIdx.x;
  __shared__ float s1[256], s2[256];
  float v[4]; float a = 0.f, b = 0.f;
  #pragma unroll
  for (int i = 0; i < 4; i++) {
    int c = tid + i * 256;
    float x = inp[(size_t)row * CC + c] + att[(size_t)row * CC + c];
    v[i] = x; a += x; b += x * x;
  }
  s1[tid] = a; s2[tid] = b;
  __syncthreads();
  for (int s = 128; s > 0; s >>= 1) {
    if (tid < s) { s1[tid] += s1[tid + s]; s2[tid] += s2[tid + s]; }
    __syncthreads();
  }
  float mean = s1[0] * (1.f / CC);
  float var  = s2[0] * (1.f / CC) - mean * mean;
  float rs   = rsqrtf(var + 1e-5f);
  #pragma unroll
  for (int i = 0; i < 4; i++) {
    int c = tid + i * 256;
    hout[(size_t)row * CC + c] = f2bf((v[i] - mean) * rs * nw[c] + nb[c]);
  }
}

// ------------------------------ out = layernorm(silu(f + bf)) -> fp32 d_out
__global__ __launch_bounds__(256) void ffn_ln_kernel(
    const float* __restrict__ fbuf, const float* __restrict__ bias,
    const float* __restrict__ lw, const float* __restrict__ lb,
    float* __restrict__ outp) {
  int row = blockIdx.x, tid = threadIdx.x;
  __shared__ float s1[256], s2[256];
  float v[4]; float a = 0.f, b = 0.f;
  #pragma unroll
  for (int i = 0; i < 4; i++) {
    int c = tid + i * 256;
    float x = fbuf[(size_t)row * CC + c] + bias[c];
    x = x / (1.f + expf(-x));           // silu
    v[i] = x; a += x; b += x * x;
  }
  s1[tid] = a; s2[tid] = b;
  __syncthreads();
  for (int s = 128; s > 0; s >>= 1) {
    if (tid < s) { s1[tid] += s1[tid + s]; s2[tid] += s2[tid + s]; }
    __syncthreads();
  }
  float mean = s1[0] * (1.f / CC);
  float var  = s2[0] * (1.f / CC) - mean * mean;
  float rs   = rsqrtf(var + 1e-5f);
  #pragma unroll
  for (int i = 0; i < 4; i++) {
    int c = tid + i * 256;
    outp[(size_t)row * CC + c] = (v[i] - mean) * rs * lw[c] + lb[c];
  }
}

// ---------------------------------------------------------------- launcher
extern "C" void kernel_launch(void* const* d_in, const int* in_sizes, int n_in,
                              void* d_out, int out_size, void* d_ws, size_t ws_size,
                              hipStream_t stream) {
  (void)in_sizes; (void)n_in; (void)out_size; (void)ws_size;
  const float* inp   = (const float*)d_in[0];
  const float* mu[6];
  for (int i = 0; i < 6; i++) mu[i] = (const float*)d_in[1 + i];
  const float* Wsrc[8];
  for (int i = 0; i < 8; i++) Wsrc[i] = (const float*)d_in[7 + i];  // Wr Ww Wk Wv Wa Wg Wo Wf
  const float* w0     = (const float*)d_in[15];
  const float* a0     = (const float*)d_in[16];
  const float* k_k    = (const float*)d_in[17];
  const float* k_a    = (const float*)d_in[18];
  const float* r_k    = (const float*)d_in[19];
  const float* lnx_w  = (const float*)d_in[20];
  const float* lnx_b  = (const float*)d_in[21];
  const float* bfv    = (const float*)d_in[22];
  const float* lnf_w  = (const float*)d_in[23];
  const float* lnf_b  = (const float*)d_in[24];
  const float* norm_w = (const float*)d_in[25];
  const float* norm_b = (const float*)d_in[26];

  char* p = (char*)d_ws;
  auto alloc = [&](size_t bytes) -> char* {
    char* r = p;
    p += (bytes + 255) & ~(size_t)255;
    return r;
  };
  const size_t NW = (size_t)CC * CC;      // 1M
  const size_t NM = (size_t)MROWS * CC;   // 4M

  unsigned short* wbfT[8];                 // transposed bf16 weights [N][K]
  for (int i = 0; i < 8; i++) wbfT[i] = (unsigned short*)alloc(NW * 2);
  unsigned short* xmix[6];
  for (int i = 0; i < 6; i++) xmix[i] = (unsigned short*)alloc(NM * 2);
  float* proj[6];                          // r, w, k, v, a, g
  for (int i = 0; i < 6; i++) proj[i] = (float*)alloc(NM * 4);
  float*          kk  = (float*)alloc(NM * 4);
  unsigned short* yg  = (unsigned short*)alloc(NM * 2);
  float*          att = (float*)alloc(NM * 4);
  unsigned short* hbf = (unsigned short*)alloc(NM * 2);
  float*          fb  = (float*)alloc(NM * 4);

  // 1. weights -> bf16, transposed to [N][K]
  dim3 tgrid(CC / 32, CC / 32), tblk(32, 8);
  for (int i = 0; i < 8; i++)
    w2bfT_kernel<<<tgrid, tblk, 0, stream>>>(Wsrc[i], wbfT[i]);

  // 2. token-shift mix -> bf16 activations
  mix_kernel<<<(int)((NM + 255) / 256), 256, 0, stream>>>(
      inp, mu[0], mu[1], mu[2], mu[3], mu[4], mu[5],
      xmix[0], xmix[1], xmix[2], xmix[3], xmix[4], xmix[5]);

  // 3. six projection GEMMs (WMMA, async double-buffered)
  dim3 gg(CC / 128, MROWS / 128);
  for (int i = 0; i < 6; i++)
    gemm_bf16_kernel<<<gg, 256, 0, stream>>>(xmix[i], wbfT[i], proj[i]);

  // 4. decay / sigmoids / kk-normalize / k'
  prep_kernel<<<(MROWS * HEADS + 255) / 256, 256, 0, stream>>>(
      proj[1], proj[2], proj[4], proj[5], kk, w0, a0, k_k, k_a);

  // 5. sequential WKV7 scan + groupnorm + bonus + gate -> bf16 (y*g)
  scan_kernel<<<BB * HEADS, 64, 0, stream>>>(
      proj[0], proj[1], proj[2], proj[3], proj[4], proj[5], kk,
      r_k, lnx_w, lnx_b, yg);

  // 6. att = (y*g) @ Wo
  gemm_bf16_kernel<<<gg, 256, 0, stream>>>(yg, wbfT[6], att);

  // 7. h = layernorm(inp + att)
  ln_add_kernel<<<MROWS, 256, 0, stream>>>(inp, att, norm_w, norm_b, hbf);

  // 8. f = h @ Wf
  gemm_bf16_kernel<<<gg, 256, 0, stream>>>(hbf, wbfT[7], fb);

  // 9. out = layernorm(silu(f + bf))
  ffn_ln_kernel<<<MROWS, 256, 0, stream>>>(fb, bfv, lnf_w, lnf_b, (float*)d_out);
}
